// LSTMClassifier_87960930222752
// MI455X (gfx1250) — matrix operand order repaired and checked by reading, not compiled, not run on hardware
//
#include <hip/hip_runtime.h>
#include <stdint.h>

typedef __attribute__((ext_vector_type(16))) __bf16        v16bf;
typedef __attribute__((ext_vector_type(8)))  float         v8f;
typedef __attribute__((ext_vector_type(4)))  unsigned int  u32x4;

// global (addrspace 1) pointer types for weight loads -> global_load_b128
typedef const __attribute__((address_space(1))) uint16_t* gu16p;
typedef const __attribute__((address_space(1))) u32x4*    gvecp;

union Frag { v16bf bf; u32x4 q[2]; };   // 32 bytes: one WMMA A/B operand per lane

__device__ __forceinline__ uint16_t f2bf(float f) {
    uint32_t u = __float_as_uint(f);
    uint32_t r = u + 0x7FFFu + ((u >> 16) & 1u);   // round-to-nearest-even
    return (uint16_t)(r >> 16);
}
__device__ __forceinline__ float sigf(float x)      { return 1.0f / (1.0f + __expf(-x)); }
__device__ __forceinline__ float tanh_fast(float x) { return 2.0f * sigf(2.0f * x) - 1.0f; }

#define Bsz 256
#define Tsz 1024
#define Isz 128
#define Hsz 256
#define G4H 1024   // 4*H

// ---------------------------------------------------------------- prep: f32 -> bf16
__global__ void cvt_bf16_kernel(const float* __restrict__ src,
                                uint16_t* __restrict__ dst, int n) {
    int i = blockIdx.x * blockDim.x + threadIdx.x;
    if (i < n) dst[i] = f2bf(src[i]);
}

// ---------------------------------------------------------------- fused LSTM
// grid = 16 blocks (one per 16-batch-row tile), 1024 threads = 32 waves.
// LDS (dynamic, 76KB): gates[16][1024] f32 | h_bf[16][256] bf16 | x_bf[16][128] bf16
__global__ __launch_bounds__(1024)
void lstm_kernel(const float* __restrict__ x,
                 const uint16_t* __restrict__ Wih_bf0,  // [4H][I] bf16 row-major
                 const uint16_t* __restrict__ Whh_bf0,  // [4H][H] bf16 row-major
                 const float* __restrict__ b_ih,
                 const float* __restrict__ b_hh,
                 const float* __restrict__ W_fc,        // [H]
                 const float* __restrict__ b_fc,        // [1]
                 float* __restrict__ out)               // [B]
{
    extern __shared__ char smem[];
    float*    gates = (float*)smem;                        // 64KB
    uint16_t* h_bf  = (uint16_t*)(smem + 65536);           // 8KB
    uint16_t* x_bf  = (uint16_t*)(smem + 65536 + 8192);    // 4KB

    const int tid  = threadIdx.x;
    const int lane = tid & 31;
    const int wave = tid >> 5;
    const int b0   = blockIdx.x * 16;

    // WMMA lane mapping (16-bit operands, wave32):
    // A: lane&15 = row M, lane>>4 selects K-half {0..7,16..23} vs {8..15,24..31}
    // B: lane&15 = col N, lane>>4 selects K-half {0..7,16..23} vs {8..15,24..31}
    const int mrow = lane & 15;
    const int hi   = lane >> 4;
    const int kbA  = hi * 8;
    const int kbB  = hi * 16;

    // this wave owns gate tiles 2*wave, 2*wave+1 (64 tiles of 16 gate-columns)
    const int g0 = (2 * wave)     * 16 + mrow;
    const int g1 = (2 * wave + 1) * 16 + mrow;
    const float bias0 = b_ih[g0] + b_hh[g0];
    const float bias1 = b_ih[g1] + b_hh[g1];

    // elementwise mapping: thread owns 4 consecutive hidden cols of one batch row
    const int bl    = tid >> 6;
    const int kbase = (tid & 63) * 4;
    float c0e = 0.f, c1e = 0.f, c2e = 0.f, c3e = 0.f;
    float h0e = 0.f, h1e = 0.f, h2e = 0.f, h3e = 0.f;

    // init: h = 0, load x[:, t=0, :] as bf16
    {
        *(uint32_t*)&h_bf[bl * Hsz + kbase]     = 0u;
        *(uint32_t*)&h_bf[bl * Hsz + kbase + 2] = 0u;
        const int bx = tid >> 6;
        const int kx = (tid & 63) * 2;
        const float* xp = x + ((size_t)(b0 + bx) * Tsz) * Isz + kx;
        uint32_t pk = (uint32_t)f2bf(xp[0]) | ((uint32_t)f2bf(xp[1]) << 16);
        *(uint32_t*)&x_bf[bx * Isz + kx] = pk;
    }
    __syncthreads();

    #pragma unroll 1
    for (int t = 0; t < Tsz; ++t) {
        // Launder weight base addresses each iteration (as integers, so the
        // compiler keeps addrspace knowledge via the explicit global-pointer
        // casts below). Blocks LICM from hoisting 192 VGPRs of B-fragments
        // (previously spilled); weights stay as in-loop GLOBAL loads from L2.
        uint64_t wih_u = (uint64_t)Wih_bf0;
        uint64_t whh_u = (uint64_t)Whh_bf0;
        asm volatile("" : "+s"(wih_u), "+s"(whh_u));
        gu16p Wih = (gu16p)wih_u;
        gu16p Whh = (gu16p)whh_u;

        // ---------------- GEMM phase: gates = bias + x_t @ Wih^T + h @ Whh^T
        // kstep-outer / tile-inner: each A fragment is loaded from LDS once
        // and feeds both gate tiles' WMMAs.
        v8f acc0, acc1;
        #pragma unroll
        for (int v = 0; v < 8; ++v) { acc0[v] = bias0; acc1[v] = bias1; }

        #pragma unroll
        for (int kk = 0; kk < 4; ++kk) {            // input projection, K=128
            Frag a, b0, b1;
            const uint16_t* pa = x_bf + mrow * Isz + kk * 32 + kbA;
            a.q[0] = *(const u32x4*)(pa);
            a.q[1] = *(const u32x4*)(pa + 16);
            gu16p pb0 = Wih + (g0 * Isz + kk * 32 + kbA);
            gu16p pb1 = Wih + (g1 * Isz + kk * 32 + kbA);
            b0.q[0] = *(gvecp)(pb0);
            b0.q[1] = *(gvecp)(pb0 + 16);
            b1.q[0] = *(gvecp)(pb1);
            b1.q[1] = *(gvecp)(pb1 + 16);
            acc0 = __builtin_amdgcn_wmma_f32_16x16x32_bf16(
                       false, a.bf, false, b0.bf, (short)0, acc0, false, false);
            acc1 = __builtin_amdgcn_wmma_f32_16x16x32_bf16(
                       false, a.bf, false, b1.bf, (short)0, acc1, false, false);
        }
        #pragma unroll
        for (int kk = 0; kk < 8; ++kk) {            // recurrent projection, K=256
            Frag a, b0, b1;
            const uint16_t* pa = h_bf + mrow * Hsz + kk * 32 + kbA;
            a.q[0] = *(const u32x4*)(pa);
            a.q[1] = *(const u32x4*)(pa + 16);
            gu16p pb0 = Whh + (g0 * Hsz + kk * 32 + kbA);
            gu16p pb1 = Whh + (g1 * Hsz + kk * 32 + kbA);
            b0.q[0] = *(gvecp)(pb0);
            b0.q[1] = *(gvecp)(pb0 + 16);
            b1.q[0] = *(gvecp)(pb1);
            b1.q[1] = *(gvecp)(pb1 + 16);
            acc0 = __builtin_amdgcn_wmma_f32_16x16x32_bf16(
                       false, a.bf, false, b0.bf, (short)0, acc0, false, false);
            acc1 = __builtin_amdgcn_wmma_f32_16x16x32_bf16(
                       false, a.bf, false, b1.bf, (short)0, acc1, false, false);
        }

        #pragma unroll
        for (int v = 0; v < 8; ++v) {               // C layout: M = v + 8*hi, N = lane&15
            gates[(v + 8 * hi) * G4H + g0] = acc0[v];
            gates[(v + 8 * hi) * G4H + g1] = acc1[v];
        }
        __syncthreads();

        // ---------------- gating phase (PyTorch order i,f,g,o)
        {
            const float* gr = gates + bl * G4H + kbase;
            float4 gi = *(const float4*)(gr);
            float4 gf = *(const float4*)(gr + Hsz);
            float4 gg = *(const float4*)(gr + 2 * Hsz);
            float4 go = *(const float4*)(gr + 3 * Hsz);

            float i0 = sigf(gi.x), f0 = sigf(gf.x), n0 = tanh_fast(gg.x), o0 = sigf(go.x);
            float i1 = sigf(gi.y), f1 = sigf(gf.y), n1 = tanh_fast(gg.y), o1 = sigf(go.y);
            float i2 = sigf(gi.z), f2 = sigf(gf.z), n2 = tanh_fast(gg.z), o2 = sigf(go.z);
            float i3 = sigf(gi.w), f3 = sigf(gf.w), n3 = tanh_fast(gg.w), o3 = sigf(go.w);

            c0e = f0 * c0e + i0 * n0;  h0e = o0 * tanh_fast(c0e);
            c1e = f1 * c1e + i1 * n1;  h1e = o1 * tanh_fast(c1e);
            c2e = f2 * c2e + i2 * n2;  h2e = o2 * tanh_fast(c2e);
            c3e = f3 * c3e + i3 * n3;  h3e = o3 * tanh_fast(c3e);

            uint32_t pk0 = (uint32_t)f2bf(h0e) | ((uint32_t)f2bf(h1e) << 16);
            uint32_t pk1 = (uint32_t)f2bf(h2e) | ((uint32_t)f2bf(h3e) << 16);
            *(uint32_t*)&h_bf[bl * Hsz + kbase]     = pk0;
            *(uint32_t*)&h_bf[bl * Hsz + kbase + 2] = pk1;
        }

        // stage x for next timestep
        if (t + 1 < Tsz) {
            const int bx = tid >> 6;
            const int kx = (tid & 63) * 2;
            const float* xp = x + ((size_t)(b0 + bx) * Tsz + (t + 1)) * Isz + kx;
            uint32_t pk = (uint32_t)f2bf(xp[0]) | ((uint32_t)f2bf(xp[1]) << 16);
            *(uint32_t*)&x_bf[bx * Isz + kx] = pk;
        }
        __syncthreads();
    }

    // ---------------- final FC: out[b] = h_T[b,:] . W_fc + b_fc  (O=1)
    float part = h0e * W_fc[kbase]     + h1e * W_fc[kbase + 1]
               + h2e * W_fc[kbase + 2] + h3e * W_fc[kbase + 3];
    float* red = gates;                 // reuse LDS
    red[tid] = part;
    __syncthreads();
    if (tid < 16) {
        float s = 0.f;
        #pragma unroll 8
        for (int j = 0; j < 64; ++j) s += red[tid * 64 + j];
        out[b0 + tid] = s + b_fc[0];
    }
}

// ---------------------------------------------------------------- launch
extern "C" void kernel_launch(void* const* d_in, const int* in_sizes, int n_in,
                              void* d_out, int out_size, void* d_ws, size_t ws_size,
                              hipStream_t stream) {
    const float* x    = (const float*)d_in[0];
    const float* W_ih = (const float*)d_in[1];
    const float* W_hh = (const float*)d_in[2];
    const float* b_ih = (const float*)d_in[3];
    const float* b_hh = (const float*)d_in[4];
    const float* W_fc = (const float*)d_in[5];
    const float* b_fc = (const float*)d_in[6];
    float* out = (float*)d_out;

    uint16_t* Wih_bf = (uint16_t*)d_ws;                 // 4H*I = 131072 elems
    uint16_t* Whh_bf = Wih_bf + (size_t)G4H * Isz;      // 4H*H = 262144 elems

    {
        int n1 = G4H * Isz;
        cvt_bf16_kernel<<<(n1 + 255) / 256, 256, 0, stream>>>(W_ih, Wih_bf, n1);
        int n2 = G4H * Hsz;
        cvt_bf16_kernel<<<(n2 + 255) / 256, 256, 0, stream>>>(W_hh, Whh_bf, n2);
    }

    const size_t lds_bytes = 65536 + 8192 + 4096;       // 76KB < 320KB WGP LDS
    lstm_kernel<<<Bsz / 16, 1024, lds_bytes, stream>>>(
        x, Wih_bf, Whh_bf, b_ih, b_hh, W_fc, b_fc, out);
    (void)in_sizes; (void)n_in; (void)out_size; (void)ws_size;
}
